// CRFLoss_6536940224946
// MI455X (gfx1250) — compile-verified, matrix operand order checked by reading, not run on hardware
//
#include <hip/hip_runtime.h>
#include <hip/hip_bf16.h>
#include <math.h>

typedef __attribute__((ext_vector_type(16))) __bf16 v16bf;
typedef __attribute__((ext_vector_type(8)))  float  v8f;

#define B_TOT 128
#define T_LEN 256
#define KDIM  256
#define MTILE 16
#define WAVES 16

union AFrag {
    v16bf v;
    uint4 q[2];
};

__global__ __launch_bounds__(512) void crf_fwd_kernel(
    const float* __restrict__ feats,
    const float* __restrict__ trans,
    const int*   __restrict__ tags,
    const int*   __restrict__ lens_g,
    float*       __restrict__ out)
{
    __shared__ float  state[MTILE][KDIM];
    __shared__ __align__(16) __bf16 p_hi[MTILE][KDIM];
    __shared__ __align__(16) __bf16 p_lo[MTILE][KDIM];
    __shared__ float  rowmax[MTILE];
    __shared__ int    lens[MTILE];

    const int tid   = threadIdx.x;
    const int wave  = tid >> 5;
    const int lane  = tid & 31;
    const int b0    = blockIdx.x * MTILE;
    const int n     = lane & 15;          // WMMA N (and A-row M) index for this lane
    const int khalf = (lane >> 4) * 8;    // K sub-offset for hi-half lanes
    const int j0    = wave * 16;          // output-column tile owned by this wave

    if (tid < MTILE) lens[tid] = lens_g[b0 + tid];

    // ---- Preload B fragments: E = exp(transitions), bf16 hi/lo split.
    // Wave w holds columns j0..j0+15 of E for all 256 K rows, in registers.
    v16bf bh[8], bl[8];
#pragma unroll
    for (int c = 0; c < 8; ++c) {
#pragma unroll
        for (int i = 0; i < 16; ++i) {
            // 16-bit B-matrix layout (mirrors A): lanes 0-15 K=kb..kb+7 then
            // kb+16..kb+23; lanes 16-31 shifted by +8.
            int k = 32 * c + khalf + (i < 8 ? i : i + 8);
            float ev = __expf(trans[k * KDIM + j0 + n]);
            __bf16 h = (__bf16)ev;
            bh[c][i] = h;
            bl[c][i] = (__bf16)(ev - (float)h);
        }
    }

    __syncthreads();  // lens visible

    // ---- state0 = mask[0] ? feats[t=0] : 0  (wave w owns batch row w)
    {
        const int b   = b0 + wave;
        const int len = lens[wave];
        const size_t base = (size_t)b * T_LEN * KDIM;
#pragma unroll
        for (int kk = 0; kk < 8; ++kk) {
            int k = lane * 8 + kk;
            float v = feats[base + k];
            state[wave][k] = (0 < len) ? v : 0.0f;
        }
    }

    // ---- serial time recurrence
    for (int t = 1; t < T_LEN; ++t) {
        __syncthreads();  // state from previous step stable

        // row max + p = exp(state - m), bf16 hi/lo split (wave w -> row w)
        float mv = -INFINITY;
        for (int k = lane; k < KDIM; k += 32) mv = fmaxf(mv, state[wave][k]);
#pragma unroll
        for (int off = 16; off > 0; off >>= 1) mv = fmaxf(mv, __shfl_xor(mv, off));
        if (lane == 0) rowmax[wave] = mv;
#pragma unroll
        for (int kk = 0; kk < 8; ++kk) {
            int k = lane * 8 + kk;
            float x = __expf(state[wave][k] - mv);
            __bf16 h = (__bf16)x;
            p_hi[wave][k] = h;
            p_lo[wave][k] = (__bf16)(x - (float)h);
        }
        __syncthreads();  // p_hi/p_lo and rowmax visible

        // D[16x16] = P[16x256] x E[256x16] via 8 chunks x 3 split-WMMAs
        v8f acc = {0.f, 0.f, 0.f, 0.f, 0.f, 0.f, 0.f, 0.f};
#pragma unroll
        for (int c = 0; c < 8; ++c) {
            const int kb = 32 * c + khalf;
            AFrag ah, al;
            ah.q[0] = *(const uint4*)&p_hi[n][kb];
            ah.q[1] = *(const uint4*)&p_hi[n][kb + 16];
            al.q[0] = *(const uint4*)&p_lo[n][kb];
            al.q[1] = *(const uint4*)&p_lo[n][kb + 16];
            acc = __builtin_amdgcn_wmma_f32_16x16x32_bf16(
                      false, ah.v, false, bh[c], (short)0, acc, false, false);
            acc = __builtin_amdgcn_wmma_f32_16x16x32_bf16(
                      false, ah.v, false, bl[c], (short)0, acc, false, false);
            acc = __builtin_amdgcn_wmma_f32_16x16x32_bf16(
                      false, al.v, false, bh[c], (short)0, acc, false, false);
        }

        // new_state = f[t] + m + log(acc), masked by t < len
        const int mhalf = (lane >> 4) * 8;
        const int j = j0 + n;
#pragma unroll
        for (int r = 0; r < 8; ++r) {
            const int M = r + mhalf;
            const int b = b0 + M;
            float fv  = feats[((size_t)b * T_LEN + t) * KDIM + j];
            float ns  = fv + rowmax[M] + __logf(acc[r]);
            bool  mk  = t < lens[M];
            float old = state[M][j];
            state[M][j] = mk ? ns : old;
        }
    }

    __syncthreads();

    // ---- final scores: wave w handles batch b0+w
    {
        const int b   = b0 + wave;
        const int len = lens[wave];

        float mv = -INFINITY;
        for (int k = lane; k < KDIM; k += 32) mv = fmaxf(mv, state[wave][k]);
#pragma unroll
        for (int off = 16; off > 0; off >>= 1) mv = fmaxf(mv, __shfl_xor(mv, off));
        float sm = 0.f;
        for (int k = lane; k < KDIM; k += 32) sm += __expf(state[wave][k] - mv);
#pragma unroll
        for (int off = 16; off > 0; off >>= 1) sm += __shfl_xor(sm, off);
        float fsc = mv + __logf(sm);

        // gold sentence score
        const int* tg = tags + (size_t)b * T_LEN;
        float us = 0.f;
        for (int t = lane; t < T_LEN; t += 32) {
            if (t < len)
                us += feats[((size_t)b * T_LEN + t) * KDIM + tg[t]];
            if (t + 1 < len)
                us += trans[tg[t] * KDIM + tg[t + 1]];
        }
#pragma unroll
        for (int off = 16; off > 0; off >>= 1) us += __shfl_xor(us, off);

        if (lane == 0) out[b] = ((len > 0) ? fsc : 0.0f) - us;
    }
}

extern "C" void kernel_launch(void* const* d_in, const int* in_sizes, int n_in,
                              void* d_out, int out_size, void* d_ws, size_t ws_size,
                              hipStream_t stream) {
    (void)in_sizes; (void)n_in; (void)out_size; (void)d_ws; (void)ws_size;
    const float* feats = (const float*)d_in[0];
    const float* trans = (const float*)d_in[1];
    const int*   tags  = (const int*)d_in[2];
    const int*   lens  = (const int*)d_in[3];
    float* out = (float*)d_out;

    dim3 grid(B_TOT / MTILE);      // 8 workgroups, one per 16-batch tile
    dim3 block(WAVES * 32);        // 16 wave32 waves
    crf_fwd_kernel<<<grid, block, 0, stream>>>(feats, trans, tags, lens, out);
}
